// ImageArm_25503515804042
// MI455X (gfx1250) — matrix operand order repaired and checked by reading, not compiled, not run on hardware
//
#include <hip/hip_runtime.h>
#include <hip/hip_bf16.h>

typedef _Float16 v16h __attribute__((ext_vector_type(16)));
typedef _Float16 v8h  __attribute__((ext_vector_type(8)));
typedef float    v8f  __attribute__((ext_vector_type(8)));

#define HW   262144   // 512*512
#define PEXP 16384    // pixels per expert
#define HID  64

// One block: 1 expert, 1 row-tile (128 pixels), 1 channel. 256 threads = 8 waves.
// Wave w handles pixels [tile*128 + w*16, +16) of the expert's index list.
__global__ __launch_bounds__(256)
void moe_arm_wmma_kernel(const float* __restrict__ image,
                         const float* __restrict__ synth,
                         const int*   __restrict__ eidx,
                         const float* __restrict__ W0, const float* __restrict__ b0,
                         const float* __restrict__ Wr, const float* __restrict__ br,
                         const float* __restrict__ Wo, const float* __restrict__ bo,
                         float* __restrict__ out,
                         int c, int in_c)
{
    // ---- LDS ----
    __shared__ __align__(32) _Float16 sF[128 * 128];   // 32 KB feature tile [pixel][K]
    __shared__ __align__(32) _Float16 sW0[16 * 512];   // 16 KB: 16 frags (nt*4+kt), frag-lane packed
    __shared__ __align__(32) _Float16 sWr[8 * 512];    //  8 KB: 8 frags (nt*2+kt)
    __shared__ __align__(32) _Float16 sWo[2 * 512];    //  2 KB: 2 frags (kt)

    const int bx   = blockIdx.x;
    const int e    = bx >> 7;          // expert
    const int tile = bx & 127;         // row-tile within expert
    const int lane = threadIdx.x & 31;
    const int wv   = threadIdx.x >> 5;

    // ================= Stage 1: build feature tile (cooperative) =================
    // thread t: pixel pi = t/2, features [half*64, half*64+64)
    {
        const int pi    = threadIdx.x >> 1;
        const int jbase = (threadIdx.x & 1) * 64;
        const int pix   = eidx[e * PEXP + tile * 128 + pi];
        const int y = pix >> 9, x = pix & 511;
        for (int jj = 0; jj < 64; ++jj) {
            const int j = jbase + jj;
            float v = 0.0f;
            if (j < 96) {
                const int t  = j / 3;
                const int ch = j - 3 * t;
                const int ii = 8 + t;                 // causal 9x9 window indices 8..39
                const int dy = ii / 9 - 4;
                const int dx = ii % 9 - 4;
                const int yy = y + dy, xx = x + dx;
                if (yy >= 0 && xx >= 0 && xx < 512)
                    v = image[(ch * 512 + yy) * 512 + xx];
            } else if (j < 102) {
                v = synth[(j - 96) * HW + pix];
            } else if (j < 105) {
                v = image[(j - 102) * HW + pix];      // weights beyond in_c are zero-padded
            }
            sF[pi * 128 + j] = (_Float16)v;
        }
    }

    // ================= Stage 2: pack weights into WMMA-B fragment layout =========
    // B frag (32x16 f16): lane L -> col n=L%16, K rows kb..kb+7 and kb+16..kb+23 with kb = kt*32 + (L<16?0:8)
    for (int task = threadIdx.x; task < 832; task += 256) {   // 26 frags * 32 lanes
        const int L   = task & 31;
        const int f   = task >> 5;
        const int n   = L & 15;
        const int kbL = (L < 16) ? 0 : 8;
        if (f < 16) {                                         // W0: K=128 (padded), N=64
            const int kt = f & 3, nt = f >> 2;
            const int kb = kt * 32 + kbL;
            const int col = nt * 16 + n;
            _Float16* dst = &sW0[(f * 32 + L) * 16];
            #pragma unroll
            for (int t = 0; t < 16; ++t) {
                const int k = kb + (t < 8 ? t : t + 8);
                const float v = (k < in_c) ? W0[(e * in_c + k) * HID + col] : 0.0f;
                dst[t] = (_Float16)v;
            }
        } else if (f < 24) {                                  // Wr: 64x64
            const int g = f - 16;
            const int kt = g & 1, nt = g >> 1;
            const int kb = kt * 32 + kbL;
            const int col = nt * 16 + n;
            _Float16* dst = &sWr[(g * 32 + L) * 16];
            #pragma unroll
            for (int t = 0; t < 16; ++t)
                dst[t] = (_Float16)Wr[(e * HID + kb + (t < 8 ? t : t + 8)) * HID + col];
        } else {                                              // Wo: 64x4 padded to 64x16
            const int kt = f - 24;
            const int kb = kt * 32 + kbL;
            _Float16* dst = &sWo[(kt * 32 + L) * 16];
            #pragma unroll
            for (int t = 0; t < 16; ++t) {
                const float v = (n < 4) ? Wo[(e * HID + kb + (t < 8 ? t : t + 8)) * 4 + n] : 0.0f;
                dst[t] = (_Float16)v;
            }
        }
    }

    __syncthreads();

    // ================= Per-wave WMMA pipeline ===================================
    const int m   = lane & 15;       // A row / C col index
    const int hi  = lane >> 4;
    const int hi8 = hi * 8;

    const _Float16* rowA = &sF[(wv * 16 + m) * 128];
    _Float16* sH = &sF[wv * 16 * 128];                 // wave-private h stage (16x64 f16)
    float* sRaw  = (float*)(&sF[wv * 16 * 128 + 1024]); // wave-private raw stage (16x4 f32)

    v8f acc[4];
    #pragma unroll
    for (int nt = 0; nt < 4; ++nt)
        acc[nt] = (v8f){0.f, 0.f, 0.f, 0.f, 0.f, 0.f, 0.f, 0.f};

    // ---- Layer 0: [16 x 128] @ [128 x 64] ----
    #pragma unroll
    for (int nt = 0; nt < 4; ++nt) {
        #pragma unroll
        for (int kt = 0; kt < 4; ++kt) {
            const int k0 = kt * 32;
            v8h lo = *(const v8h*)(rowA + k0 + hi8);
            v8h hv = *(const v8h*)(rowA + k0 + 16 + hi8);
            v16h a = __builtin_shufflevector(lo, hv, 0,1,2,3,4,5,6,7,8,9,10,11,12,13,14,15);
            v16h b = *(const v16h*)(&sW0[((nt * 4 + kt) * 32 + lane) * 16]);
            acc[nt] = __builtin_amdgcn_wmma_f32_16x16x32_f16(
                false, a, false, b, (short)0, acc[nt], false, false);
        }
    }
    #pragma unroll
    for (int nt = 0; nt < 4; ++nt) {
        const float bv = b0[e * HID + nt * 16 + m];
        #pragma unroll
        for (int i = 0; i < 8; ++i)
            acc[nt][i] = fmaxf(acc[nt][i] + bv, 0.0f);
    }

    // ---- stage h0 into wave-private LDS (f16, row-major 16x64) ----
    asm volatile("s_wait_dscnt 0x0" ::: "memory");
    #pragma unroll
    for (int nt = 0; nt < 4; ++nt)
        #pragma unroll
        for (int i = 0; i < 8; ++i)
            sH[(i + hi8) * HID + nt * 16 + m] = (_Float16)acc[nt][i];
    asm volatile("s_wait_dscnt 0x0" ::: "memory");

    // ---- Residual layer: h1 = relu(h0 @ Wr + br + h0); C init = h0 (free residual) ----
    #pragma unroll
    for (int nt = 0; nt < 4; ++nt) {
        v8f c2 = acc[nt];
        #pragma unroll
        for (int kt = 0; kt < 2; ++kt) {
            const int k0 = kt * 32;
            v8h lo = *(const v8h*)(sH + m * HID + k0 + hi8);
            v8h hv = *(const v8h*)(sH + m * HID + k0 + 16 + hi8);
            v16h a = __builtin_shufflevector(lo, hv, 0,1,2,3,4,5,6,7,8,9,10,11,12,13,14,15);
            v16h b = *(const v16h*)(&sWr[((nt * 2 + kt) * 32 + lane) * 16]);
            c2 = __builtin_amdgcn_wmma_f32_16x16x32_f16(
                false, a, false, b, (short)0, c2, false, false);
        }
        const float bv = br[e * HID + nt * 16 + m];
        #pragma unroll
        for (int i = 0; i < 8; ++i)
            acc[nt][i] = fmaxf(c2[i] + bv, 0.0f);
    }

    // ---- stage h1 ----
    asm volatile("s_wait_dscnt 0x0" ::: "memory");
    #pragma unroll
    for (int nt = 0; nt < 4; ++nt)
        #pragma unroll
        for (int i = 0; i < 8; ++i)
            sH[(i + hi8) * HID + nt * 16 + m] = (_Float16)acc[nt][i];
    asm volatile("s_wait_dscnt 0x0" ::: "memory");

    // ---- Output layer: [16 x 64] @ [64 x 16(pad of 4)] ----
    v8f accO = (v8f){0.f, 0.f, 0.f, 0.f, 0.f, 0.f, 0.f, 0.f};
    #pragma unroll
    for (int kt = 0; kt < 2; ++kt) {
        const int k0 = kt * 32;
        v8h lo = *(const v8h*)(sH + m * HID + k0 + hi8);
        v8h hv = *(const v8h*)(sH + m * HID + k0 + 16 + hi8);
        v16h a = __builtin_shufflevector(lo, hv, 0,1,2,3,4,5,6,7,8,9,10,11,12,13,14,15);
        v16h b = *(const v16h*)(&sWo[(kt * 32 + lane) * 16]);
        accO = __builtin_amdgcn_wmma_f32_16x16x32_f16(
            false, a, false, b, (short)0, accO, false, false);
    }
    const float bov = (m < 4) ? bo[e * 4 + m] : 0.0f;
    #pragma unroll
    for (int i = 0; i < 8; ++i) accO[i] += bov;

    // ---- stage raw (only cols 0..3 are real) ----
    asm volatile("s_wait_dscnt 0x0" ::: "memory");
    if (m < 4) {
        #pragma unroll
        for (int i = 0; i < 8; ++i)
            sRaw[(i + hi8) * 4 + m] = accO[i];
    }
    asm volatile("s_wait_dscnt 0x0" ::: "memory");

    // ---- gate + add synth, scatter to output (lanes 0..15, one pixel each) ----
    if (lane < 16) {
        const float p0 = sRaw[lane * 4 + 0];
        const float p1 = sRaw[lane * 4 + 1];
        const float g0 = sRaw[lane * 4 + 2];
        const float g1 = sRaw[lane * 4 + 3];
        const float gated0 = p0 * (1.0f / (1.0f + __expf(-g0)));
        const float gated1 = p1 * (1.0f / (1.0f + __expf(-g1)));
        const int pix = eidx[e * PEXP + tile * 128 + wv * 16 + lane];
        const int ch0 = 2 * c;
        out[(ch0 + 0) * HW + pix] = synth[(ch0 + 0) * HW + pix] + gated0;
        out[(ch0 + 1) * HW + pix] = synth[(ch0 + 1) * HW + pix] + gated1;
    }
}

extern "C" void kernel_launch(void* const* d_in, const int* in_sizes, int n_in,
                              void* d_out, int out_size, void* d_ws, size_t ws_size,
                              hipStream_t stream) {
    const float* image = (const float*)d_in[0];
    const float* synth = (const float*)d_in[1];
    const int*   eidx  = (const int*)d_in[2];
    float* out = (float*)d_out;

    for (int c = 0; c < 3; ++c) {
        const float* W0 = (const float*)d_in[3 + 6 * c + 0];
        const float* b0 = (const float*)d_in[3 + 6 * c + 1];
        const float* Wr = (const float*)d_in[3 + 6 * c + 2];
        const float* br = (const float*)d_in[3 + 6 * c + 3];
        const float* Wo = (const float*)d_in[3 + 6 * c + 4];
        const float* bo = (const float*)d_in[3 + 6 * c + 5];
        moe_arm_wmma_kernel<<<dim3(16 * 128), dim3(256), 0, stream>>>(
            image, synth, eidx, W0, b0, Wr, br, Wo, bo, out, c, 102 + c);
    }
}
